// WaveNetNvidia_1185410974036
// MI455X (gfx1250) — matrix operand order repaired and checked by reading, not compiled, not run on hardware
//
#include <hip/hip_runtime.h>
#include <math.h>

// ---------------- problem constants ----------------
#define N_LAYERS 12
#define CCH      256        // residual channels
#define CMEL     640        // conditioning channels
#define TLEN     4096
#define MROWS    16384      // B * T
#define KIN      768        // 3 * CCH (im2col K of dilated conv)
#define KTOT     1408       // KIN + CMEL (fused in+cond GEMM)
#define LDSK     40         // padded LDS row stride (32 data bf16 + 8 pad)

typedef __attribute__((ext_vector_type(16))) __bf16 v16bf;
typedef __attribute__((ext_vector_type(8)))  float  v8f;

union FragU { uint4 q[2]; v16bf v; };

// ---------------- CDNA5 async global->LDS path (guarded) ----------------
#if defined(__has_builtin)
#if __has_builtin(__builtin_amdgcn_global_load_async_to_lds_b128)
#define HAVE_ASYNC_LDS 1
#endif
#endif
#ifndef HAVE_ASYNC_LDS
#define HAVE_ASYNC_LDS 0
#endif

#if HAVE_ASYNC_LDS
// Exact parameter types per hipcc diagnostic: pointer to vector_size(16) int,
// src in AS1 (printed as __device__), dst in AS3 (LDS).
typedef int v4i_vs __attribute__((vector_size(4 * sizeof(int))));
typedef __attribute__((address_space(1))) v4i_vs* gas1_v4i;
typedef __attribute__((address_space(3))) v4i_vs* las3_v4i;
#endif

__device__ __forceinline__ void copy16(const unsigned short* g, unsigned short* l) {
#if HAVE_ASYNC_LDS
  // global generic addr == AS1 addr; LDS generic addr low 32 bits == LDS offset
  __builtin_amdgcn_global_load_async_to_lds_b128(
      (gas1_v4i)(unsigned long long)(uintptr_t)g,
      (las3_v4i)(unsigned)(uintptr_t)l,
      0, 0);
#else
  *(uint4*)l = *(const uint4*)g;
#endif
}

__device__ __forceinline__ void zero16(unsigned short* l) {
  uint4 z = {0u, 0u, 0u, 0u};
  *(uint4*)l = z;
}

__device__ __forceinline__ void wait_async() {
#if HAVE_ASYNC_LDS
#if __has_builtin(__builtin_amdgcn_s_wait_asynccnt)
  __builtin_amdgcn_s_wait_asynccnt(0);
#else
  asm volatile("s_wait_asynccnt 0" ::: "memory");
#endif
#endif
}

__device__ __forceinline__ unsigned short f32_to_bf16_raw(float f) {
  union { float f; unsigned u; } c; c.f = f;
  unsigned u = c.u;
  unsigned r = 0x7FFFu + ((u >> 16) & 1u);   // round-to-nearest-even
  return (unsigned short)((u + r) >> 16);
}

__device__ __forceinline__ v8f wmma_bf16(v16bf a, v16bf b, v8f c) {
  return __builtin_amdgcn_wmma_f32_16x16x32_bf16(false, a, false, b, (short)0, c,
                                                 false, false);
}

// ---------------- weight-norm prep: normalized, transposed bf16 weights ----------------
__global__ void prep_weights_kernel(
    const float* __restrict__ v_start, const float* __restrict__ g_start,
    const float* __restrict__ v_cond,  const float* __restrict__ g_cond,
    const float* __restrict__ v_in,    const float* __restrict__ g_in,
    const float* __restrict__ v_rs,    const float* __restrict__ g_rs,
    unsigned short* __restrict__ Wc_t,  // [6144, 640]
    unsigned short* __restrict__ Wi_t,  // [12*512, 768]
    unsigned short* __restrict__ Wr_t,  // [12*512, 256]
    float* __restrict__ Wstart_n)       // [4, 256]
{
  int id = blockIdx.x * blockDim.x + threadIdx.x;
  if (id < 6144) {
    int n = id;
    float ss = 0.f;
    for (int k = 0; k < CMEL; ++k) { float v = v_cond[(size_t)k * 6144 + n]; ss += v * v; }
    float inv = g_cond[n] / sqrtf(fmaxf(ss, 1e-12f));
    for (int k = 0; k < CMEL; ++k)
      Wc_t[(size_t)n * CMEL + k] = f32_to_bf16_raw(v_cond[(size_t)k * 6144 + n] * inv);
  } else if (id < 12288) {
    int r = id - 6144, l = r >> 9, n = r & 511;
    float ss = 0.f;
    for (int k = 0; k < KIN; ++k) { float v = v_in[((size_t)l * KIN + k) * 512 + n]; ss += v * v; }
    float inv = g_in[l * 512 + n] / sqrtf(fmaxf(ss, 1e-12f));
    for (int k = 0; k < KIN; ++k)
      Wi_t[((size_t)l * 512 + n) * KIN + k] =
          f32_to_bf16_raw(v_in[((size_t)l * KIN + k) * 512 + n] * inv);
  } else if (id < 18432) {
    int r = id - 12288, l = r >> 9, n = r & 511;
    float ss = 0.f;
    for (int k = 0; k < CCH; ++k) { float v = v_rs[((size_t)l * CCH + k) * 512 + n]; ss += v * v; }
    float inv = g_rs[l * 512 + n] / sqrtf(fmaxf(ss, 1e-12f));
    for (int k = 0; k < CCH; ++k)
      Wr_t[((size_t)l * 512 + n) * CCH + k] =
          f32_to_bf16_raw(v_rs[((size_t)l * CCH + k) * 512 + n] * inv);
  } else if (id < 18688) {
    int c = id - 18432;
    float ss = 0.f;
    for (int j = 0; j < 4; ++j) { float v = v_start[j * 256 + c]; ss += v * v; }
    float inv = g_start[c] / sqrtf(fmaxf(ss, 1e-12f));
    for (int j = 0; j < 4; ++j) Wstart_n[j * 256 + c] = v_start[j * 256 + c] * inv;
  }
}

// ---------------- start 1x1 conv (K=4) ----------------
__global__ void start_conv_kernel(
    const float* __restrict__ audio, const float* __restrict__ Wstart_n,
    const float* __restrict__ b_start,
    float* __restrict__ started, unsigned short* __restrict__ started_bf16)
{
  int idx = blockIdx.x * 256 + threadIdx.x;       // exactly MROWS*CCH threads
  int m = idx >> 8, c = idx & 255;
  float s = b_start[c];
#pragma unroll
  for (int j = 0; j < 4; ++j) s += audio[m * 4 + j] * Wstart_n[j * 256 + c];
  started[idx] = s;
  started_bf16[idx] = f32_to_bf16_raw(s);
}

__global__ void spect_cvt_kernel(const float* __restrict__ spect,
                                 unsigned short* __restrict__ spect_bf16)
{
  size_t idx = (size_t)blockIdx.x * 256 + threadIdx.x;   // MROWS*CMEL threads
  spect_bf16[idx] = f32_to_bf16_raw(spect[idx]);
}

// ---------------- fused dilated-conv + cond GEMM (WMMA bf16, async double-buffer) ----------------
__global__ __launch_bounds__(256) void layer_in_gemm(
    const unsigned short* __restrict__ started_bf16,   // [M, 256]
    const unsigned short* __restrict__ spect_bf16,     // [M, 640]
    const unsigned short* __restrict__ Wi_t,           // [12*512, 768]
    const unsigned short* __restrict__ Wc_t,           // [6144, 640]
    const float* __restrict__ b_in,                    // [12*512]
    const float* __restrict__ b_cond,                  // [6144]
    float* __restrict__ x,                             // [M, 512]
    int layer, int dil)
{
  __shared__ unsigned short ldsA[2][128 * LDSK];
  __shared__ unsigned short ldsB[2][128 * LDSK];

  const int tid  = threadIdx.x;
  const int lane = tid & 31;
  const int wave = tid >> 5;
  const int wm = wave & 1;          // 2 waves across M (64 rows each)
  const int wn = wave >> 1;         // 4 waves across N (32 cols each)
  const int lr = lane & 15;
  const int lg = lane >> 4;
  const int m0 = blockIdx.y * 128;
  const int n0 = blockIdx.x * 128;  // within [0,512)

  auto stage = [&](int kb, unsigned short* bufA, unsigned short* bufB) {
    const int kc = kb * 32;
#pragma unroll
    for (int h = 0; h < 2; ++h) {
      int u = tid + h * 256;
      int row = u >> 2, q = u & 3;
      int ks = kc + q * 8;
      // ---- A: im2col of started, then spect ----
      {
        int m = m0 + row;
        unsigned short* l = bufA + row * LDSK + q * 8;
        if (ks < KIN) {
          int tap = ks >> 8;
          int cin = ks & 255;
          int t  = m & (TLEN - 1);
          int ts = t + (tap - 1) * dil;
          if (ts >= 0 && ts < TLEN)
            copy16(started_bf16 + (size_t)(m + (tap - 1) * dil) * CCH + cin, l);
          else
            zero16(l);
        } else {
          copy16(spect_bf16 + (size_t)m * CMEL + (ks - KIN), l);
        }
      }
      // ---- B: transposed normalized weights, K contiguous ----
      {
        int n = n0 + row;
        const unsigned short* g;
        if (ks < KIN) g = Wi_t + ((size_t)layer * 512 + n) * KIN + ks;
        else          g = Wc_t + ((size_t)layer * 512 + n) * CMEL + (ks - KIN);
        __builtin_prefetch(g + 32, 0, 1);   // next K-chunk of this weight row
        copy16(g, bufB + row * LDSK + q * 8);
      }
    }
  };

  v8f acc[4][2];
  v8f zero = {0.f, 0.f, 0.f, 0.f, 0.f, 0.f, 0.f, 0.f};
#pragma unroll
  for (int f = 0; f < 4; ++f) { acc[f][0] = zero; acc[f][1] = zero; }

  const int KCH = KTOT / 32;        // 44; chunk never straddles tap/cond split
  stage(0, ldsA[0], ldsB[0]);
  wait_async();
  __syncthreads();

  for (int kb = 0; kb < KCH; ++kb) {
    unsigned short* curA = ldsA[kb & 1];
    unsigned short* curB = ldsB[kb & 1];
    if (kb + 1 < KCH) stage(kb + 1, ldsA[(kb + 1) & 1], ldsB[(kb + 1) & 1]);

    v16bf bfr[2];
#pragma unroll
    for (int j = 0; j < 2; ++j) {
      FragU fb;
      const unsigned short* bp = curB + (wn * 32 + j * 16 + lr) * LDSK + lg * 16;
      fb.q[0] = *(const uint4*)(bp);
      fb.q[1] = *(const uint4*)(bp + 8);
      bfr[j] = fb.v;
    }
#pragma unroll
    for (int f = 0; f < 4; ++f) {
      FragU fa;
      const unsigned short* ap = curA + (wm * 64 + f * 16 + lr) * LDSK + lg * 16;
      fa.q[0] = *(const uint4*)(ap);
      fa.q[1] = *(const uint4*)(ap + 8);
      acc[f][0] = wmma_bf16(fa.v, bfr[0], acc[f][0]);
      acc[f][1] = wmma_bf16(fa.v, bfr[1], acc[f][1]);
    }
    wait_async();
    __syncthreads();
  }
  // --- epilogue: bias and write x ---
#pragma unroll
  for (int f = 0; f < 4; ++f)
#pragma unroll
    for (int j = 0; j < 2; ++j) {
      int n  = n0 + wn * 32 + j * 16 + lr;
      int mb = m0 + wm * 64 + f * 16 + lg * 8;
      float bias = b_in[layer * 512 + n] + b_cond[layer * 512 + n];
#pragma unroll
      for (int v = 0; v < 8; ++v)
        x[(size_t)(mb + v) * 512 + n] = acc[f][j][v] + bias;
    }
}

// ---------------- gated activation ----------------
__global__ void gate_kernel(const float* __restrict__ x,
                            unsigned short* __restrict__ act_bf16)
{
  int idx = blockIdx.x * 256 + threadIdx.x;   // MROWS*CCH threads
  int m = idx >> 8, c = idx & 255;
  float a = x[(size_t)m * 512 + c];
  float b = x[(size_t)m * 512 + 256 + c];
  float t = tanhf(a);
  float s = 1.0f / (1.0f + __expf(-b));
  act_bf16[idx] = f32_to_bf16_raw(t * s);
}

// ---------------- res/skip GEMM (WMMA bf16, K=256, async double-buffer) ----------------
__global__ __launch_bounds__(256) void layer_rs_gemm(
    const unsigned short* __restrict__ act_bf16,  // [M, 256]
    const unsigned short* __restrict__ Wr_t,      // [12*512, 256]
    const float* __restrict__ b_rs,               // [12*512]
    float* __restrict__ started,
    unsigned short* __restrict__ started_bf16,
    float* __restrict__ skip,
    int layer, int is_last, int is_first)
{
  __shared__ unsigned short ldsA[2][128 * LDSK];
  __shared__ unsigned short ldsB[2][128 * LDSK];

  const int tid  = threadIdx.x;
  const int lane = tid & 31;
  const int wave = tid >> 5;
  const int wm = wave & 1;
  const int wn = wave >> 1;
  const int lr = lane & 15;
  const int lg = lane >> 4;
  const int m0 = blockIdx.y * 128;
  const int n0 = blockIdx.x * 128;

  auto stage = [&](int kb, unsigned short* bufA, unsigned short* bufB) {
    const int kc = kb * 32;
#pragma unroll
    for (int h = 0; h < 2; ++h) {
      int u = tid + h * 256;
      int row = u >> 2, q = u & 3;
      int ks = kc + q * 8;
      copy16(act_bf16 + (size_t)(m0 + row) * CCH + ks, bufA + row * LDSK + q * 8);
      const unsigned short* g = Wr_t + ((size_t)layer * 512 + n0 + row) * CCH + ks;
      __builtin_prefetch(g + 32, 0, 1);
      copy16(g, bufB + row * LDSK + q * 8);
    }
  };

  v8f acc[4][2];
  v8f zero = {0.f, 0.f, 0.f, 0.f, 0.f, 0.f, 0.f, 0.f};
#pragma unroll
  for (int f = 0; f < 4; ++f) { acc[f][0] = zero; acc[f][1] = zero; }

  const int KCH = CCH / 32;   // 8
  stage(0, ldsA[0], ldsB[0]);
  wait_async();
  __syncthreads();

  for (int kb = 0; kb < KCH; ++kb) {
    unsigned short* curA = ldsA[kb & 1];
    unsigned short* curB = ldsB[kb & 1];
    if (kb + 1 < KCH) stage(kb + 1, ldsA[(kb + 1) & 1], ldsB[(kb + 1) & 1]);

    v16bf bfr[2];
#pragma unroll
    for (int j = 0; j < 2; ++j) {
      FragU fb;
      const unsigned short* bp = curB + (wn * 32 + j * 16 + lr) * LDSK + lg * 16;
      fb.q[0] = *(const uint4*)(bp);
      fb.q[1] = *(const uint4*)(bp + 8);
      bfr[j] = fb.v;
    }
#pragma unroll
    for (int f = 0; f < 4; ++f) {
      FragU fa;
      const unsigned short* ap = curA + (wm * 64 + f * 16 + lr) * LDSK + lg * 16;
      fa.q[0] = *(const uint4*)(ap);
      fa.q[1] = *(const uint4*)(ap + 8);
      acc[f][0] = wmma_bf16(fa.v, bfr[0], acc[f][0]);
      acc[f][1] = wmma_bf16(fa.v, bfr[1], acc[f][1]);
    }
    wait_async();
    __syncthreads();
  }
  // --- epilogue: res into started (f32 + bf16), skip accumulate ---
#pragma unroll
  for (int f = 0; f < 4; ++f)
#pragma unroll
    for (int j = 0; j < 2; ++j) {
      int n  = n0 + wn * 32 + j * 16 + lr;
      int mb = m0 + wm * 64 + f * 16 + lg * 8;
      float bias = b_rs[layer * 512 + n];
#pragma unroll
      for (int v = 0; v < 8; ++v) {
        float rs = acc[f][j][v] + bias;
        size_t m = (size_t)(mb + v);
        if (is_last) {                       // only skip half launched (N=256)
          skip[m * CCH + n] += rs;
        } else if (n < CCH) {                // residual
          float s = started[m * CCH + n] + rs;
          started[m * CCH + n] = s;
          started_bf16[m * CCH + n] = f32_to_bf16_raw(s);
        } else {                             // skip
          int c = n - CCH;
          if (is_first) skip[m * CCH + c] = rs;
          else          skip[m * CCH + c] += rs;
        }
      }
    }
}

// ---------------- end 1x1 conv (N=8) ----------------
__global__ void end_conv_kernel(const float* __restrict__ skip,
                                const float* __restrict__ k_end,
                                const float* __restrict__ b_end,
                                float* __restrict__ out)
{
  int m = blockIdx.x * 256 + threadIdx.x;   // MROWS threads
  float a[8];
#pragma unroll
  for (int o = 0; o < 8; ++o) a[o] = b_end[o];
  for (int c = 0; c < CCH; ++c) {
    float s = skip[(size_t)m * CCH + c];
#pragma unroll
    for (int o = 0; o < 8; ++o) a[o] += s * k_end[c * 8 + o];
  }
#pragma unroll
  for (int o = 0; o < 8; ++o) out[(size_t)m * 8 + o] = a[o];
}

// ---------------- launcher ----------------
extern "C" void kernel_launch(void* const* d_in, const int* in_sizes, int n_in,
                              void* d_out, int out_size, void* d_ws, size_t ws_size,
                              hipStream_t stream) {
  const float* audio   = (const float*)d_in[0];
  const float* spect   = (const float*)d_in[1];
  const float* v_start = (const float*)d_in[2];
  const float* g_start = (const float*)d_in[3];
  const float* b_start = (const float*)d_in[4];
  const float* v_cond  = (const float*)d_in[5];
  const float* g_cond  = (const float*)d_in[6];
  const float* b_cond  = (const float*)d_in[7];
  const float* v_in    = (const float*)d_in[8];
  const float* g_in    = (const float*)d_in[9];
  const float* b_in    = (const float*)d_in[10];
  const float* v_rs    = (const float*)d_in[11];
  const float* g_rs    = (const float*)d_in[12];
  const float* b_rs    = (const float*)d_in[13];
  const float* k_end   = (const float*)d_in[14];
  const float* b_end   = (const float*)d_in[15];
  float* out = (float*)d_out;

  char* ws = (char*)d_ws;
  size_t off = 0;
  auto take = [&](size_t bytes) -> char* {
    char* p = ws + off;
    off = (off + bytes + 255) & ~(size_t)255;
    return p;
  };
  float* started          = (float*)take((size_t)MROWS * CCH * 4);
  float* xbuf             = (float*)take((size_t)MROWS * 512 * 4);
  float* skip             = (float*)take((size_t)MROWS * CCH * 4);
  float* Wstart_n         = (float*)take(4 * 256 * 4);
  unsigned short* spect16 = (unsigned short*)take((size_t)MROWS * CMEL * 2);
  unsigned short* start16 = (unsigned short*)take((size_t)MROWS * CCH * 2);
  unsigned short* act16   = (unsigned short*)take((size_t)MROWS * CCH * 2);
  unsigned short* Wc_t    = (unsigned short*)take((size_t)6144 * CMEL * 2);
  unsigned short* Wi_t    = (unsigned short*)take((size_t)N_LAYERS * 512 * KIN * 2);
  unsigned short* Wr_t    = (unsigned short*)take((size_t)N_LAYERS * 512 * CCH * 2);

  prep_weights_kernel<<<(18688 + 255) / 256, 256, 0, stream>>>(
      v_start, g_start, v_cond, g_cond, v_in, g_in, v_rs, g_rs,
      Wc_t, Wi_t, Wr_t, Wstart_n);

  start_conv_kernel<<<MROWS * CCH / 256, 256, 0, stream>>>(
      audio, Wstart_n, b_start, started, start16);
  spect_cvt_kernel<<<MROWS * CMEL / 256, 256, 0, stream>>>(spect, spect16);

  for (int l = 0; l < N_LAYERS; ++l) {
    int dil = 1 << l;
    dim3 g1(512 / 128, MROWS / 128);
    layer_in_gemm<<<g1, 256, 0, stream>>>(start16, spect16, Wi_t, Wc_t,
                                          b_in, b_cond, xbuf, l, dil);
    gate_kernel<<<MROWS * CCH / 256, 256, 0, stream>>>(xbuf, act16);
    int is_last  = (l == N_LAYERS - 1);
    int is_first = (l == 0);
    dim3 g3(is_last ? (256 / 128) : (512 / 128), MROWS / 128);
    layer_rs_gemm<<<g3, 256, 0, stream>>>(act16, Wr_t, b_rs, started, start16,
                                          skip, l, is_last, is_first);
  }

  end_conv_kernel<<<MROWS / 256, 256, 0, stream>>>(skip, k_end, b_end, out);
}
